// HRVFeatNet_22110491639879
// MI455X (gfx1250) — matrix operand order repaired and checked by reading, not compiled, not run on hardware
//
#include <hip/hip_runtime.h>
#include <hip/hip_bf16.h>

typedef __attribute__((ext_vector_type(16))) _Float16 v16h;
typedef __attribute__((ext_vector_type(8)))  float    v8f;

#define T_LEN   3000
#define NWORDS  94          // ceil(3000/32)
#define K_PEAKS 512
#define FS      100.0f
#define NFEAT   16
#define BLK     256

// ---------------- block reductions (LDS tree, wave-size agnostic) ----------------
__device__ __forceinline__ float blockReduceSum(float v, float* red, int tid) {
    red[tid] = v; __syncthreads();
    for (int o = BLK / 2; o > 0; o >>= 1) {
        if (tid < o) red[tid] += red[tid + o];
        __syncthreads();
    }
    float r = red[0]; __syncthreads();
    return r;
}
__device__ __forceinline__ float blockReduceMax(float v, float* red, int tid) {
    red[tid] = v; __syncthreads();
    for (int o = BLK / 2; o > 0; o >>= 1) {
        if (tid < o) red[tid] = fmaxf(red[tid], red[tid + o]);
        __syncthreads();
    }
    float r = red[0]; __syncthreads();
    return r;
}
__device__ __forceinline__ float blockReduceMin(float v, float* red, int tid) {
    red[tid] = v; __syncthreads();
    for (int o = BLK / 2; o > 0; o >>= 1) {
        if (tid < o) red[tid] = fminf(red[tid], red[tid + o]);
        __syncthreads();
    }
    float r = red[0]; __syncthreads();
    return r;
}

// ---------------- kernel 1: per-row HRV features ----------------
// Row (12 KB) is staged into LDS once via CDNA5 async global->LDS DMA
// (GLOBAL_LOAD_ASYNC_TO_LDS_B32, ASYNCcnt), then all passes (stats, peak
// detect with +/-1 neighbors, amplitude gather) are served from LDS.
__global__ __launch_bounds__(BLK) void feat_kernel(const float* __restrict__ x,
                                                   float* __restrict__ f) {
    const int row = blockIdx.x;
    const int tid = threadIdx.x;
    const float* __restrict__ xr = x + (size_t)row * T_LEN;

    __shared__ float    xs[T_LEN];
    __shared__ float    red[BLK];
    __shared__ unsigned pkmask[NWORDS];
    __shared__ int      wcnt[NWORDS + 1];
    __shared__ int      pk[K_PEAKS];
    __shared__ float    rr[K_PEAKS];
    __shared__ int      s_npk;

    // ---- stage row into LDS with async DMA loads ----
    for (int t = tid; t < T_LEN; t += BLK) {
        unsigned     ldsOff = (unsigned)(unsigned long long)(&xs[t]);  // low 32b = LDS addr
        const float* gp     = xr + t;
        asm volatile("global_load_async_to_lds_b32 %0, %1, off"
                     : : "v"(ldsOff), "v"(gp) : "memory");
    }
    asm volatile("s_wait_asynccnt 0x0" ::: "memory");   // this wave's DMAs landed
    __syncthreads();                                    // all waves' DMAs visible

    // ---- pass A: signal statistics ----
    float s = 0.f, s2 = 0.f, mx = -3.4e38f, mn = 3.4e38f;
    for (int t = tid; t < T_LEN; t += BLK) {
        float v = xs[t];
        s += v; s2 += v * v;
        mx = fmaxf(mx, v); mn = fminf(mn, v);
    }
    const float S   = blockReduceSum(s,  red, tid);
    const float S2  = blockReduceSum(s2, red, tid);
    const float MX  = blockReduceMax(mx, red, tid);
    const float MN  = blockReduceMin(mn, red, tid);
    const float invT    = 1.0f / (float)T_LEN;
    const float sigMean = S * invT;
    const float meanSq  = S2 * invT;
    float var = meanSq - sigMean * sigMean;
    var = var > 0.f ? var : 0.f;
    const float sigStd = sqrtf(var);
    const float thr    = sigMean + 0.5f * sigStd;

    // ---- pass B: peak detection -> ordered compaction of first K_PEAKS peaks ----
    for (int wd = tid; wd < NWORDS; wd += BLK) pkmask[wd] = 0u;
    __syncthreads();
    for (int t = tid; t < T_LEN; t += BLK) {
        if (t >= 1 && t <= T_LEN - 2) {
            float v = xs[t];
            if (v > xs[t - 1] && v >= xs[t + 1] && v > thr)
                atomicOr(&pkmask[t >> 5], 1u << (t & 31));
        }
    }
    __syncthreads();
    if (tid < NWORDS) wcnt[tid] = __popc(pkmask[tid]);
    __syncthreads();
    if (tid == 0) {                       // exclusive scan over 94 words
        int acc = 0;
        for (int i = 0; i < NWORDS; ++i) { int c = wcnt[i]; wcnt[i] = acc; acc += c; }
        wcnt[NWORDS] = acc;
        s_npk = acc < K_PEAKS ? acc : K_PEAKS;
    }
    __syncthreads();
    if (tid < NWORDS) {                   // in-order compaction (per-word)
        unsigned m = pkmask[tid];
        int base = wcnt[tid];
        while (m) {
            int b = __ffs(m) - 1;
            m &= m - 1;
            if (base < K_PEAKS) pk[base] = (tid << 5) + b;
            ++base;
        }
    }
    __syncthreads();

    const int   n_pk   = s_npk;
    const float n_pk_f = (float)n_pk;

    // ---- amplitudes at peaks: two-pass mean/std (population, /max(n,1)) ----
    float as = 0.f;
    for (int i = tid; i < n_pk; i += BLK) as += xs[pk[i]];
    const float ampMean = blockReduceSum(as, red, tid) / fmaxf(n_pk_f, 1.0f);
    float ad = 0.f;
    for (int i = tid; i < n_pk; i += BLK) { float d = xs[pk[i]] - ampMean; ad += d * d; }
    const float ampStd = sqrtf(blockReduceSum(ad, red, tid) / fmaxf(n_pk_f, 1.0f));

    // ---- RR intervals ----
    const int   n_rr   = n_pk >= 2 ? n_pk - 1 : 0;
    const float n_rr_f = (float)n_rr;
    float rs = 0.f, rmin = 3.4e38f, rmax = -3.4e38f;
    for (int i = tid; i < n_rr; i += BLK) {
        float t0 = (float)pk[i]     / FS;      // times = idx / fs, then diff
        float t1 = (float)pk[i + 1] / FS;
        float r  = t1 - t0;
        rr[i] = r;
        rs += r; rmin = fminf(rmin, r); rmax = fmaxf(rmax, r);
    }
    const float rrSum  = blockReduceSum(rs, red, tid);
    const float rrMin  = blockReduceMin(rmin, red, tid);
    const float rrMax  = blockReduceMax(rmax, red, tid);
    const float rrMean = rrSum / fmaxf(n_rr_f, 1.0f);
    float rv = 0.f;
    for (int i = tid; i < n_rr; i += BLK) { float d = rr[i] - rrMean; rv += d * d; }
    const float sdnn = sqrtf(blockReduceSum(rv, red, tid) / fmaxf(n_rr_f, 1.0f));

    // ---- dRR ----
    const int   n_drr   = n_rr >= 2 ? n_rr - 1 : 0;
    const float n_drr_f = (float)n_drr;
    float ds2 = 0.f, dc = 0.f;
    for (int i = tid; i < n_drr; i += BLK) {
        float d = rr[i + 1] - rr[i];
        ds2 += d * d;
        dc  += (fabsf(d) > 0.05f) ? 1.0f : 0.0f;
    }
    const float rmssd = sqrtf(blockReduceSum(ds2, red, tid) / fmaxf(n_drr_f, 1.0f));
    const float pnn50 = blockReduceSum(dc, red, tid) / fmaxf(n_drr_f, 1.0f);

    if (tid == 0) {
        float* fo = f + (size_t)row * NFEAT;
        fo[0]  = n_pk_f;
        fo[1]  = rrMean;
        fo[2]  = sdnn;
        fo[3]  = rmssd;
        fo[4]  = pnn50;
        fo[5]  = n_rr > 0 ? rrMin : 0.f;
        fo[6]  = n_rr > 0 ? rrMax : 0.f;
        fo[7]  = sdnn / (rrMean + 1e-8f);
        fo[8]  = n_rr > 0 ? 60.0f / (rrMean + 1e-8f) : 0.f;
        fo[9]  = ampMean;
        fo[10] = ampStd;
        fo[11] = sigMean;
        fo[12] = sigStd;
        fo[13] = MX;
        fo[14] = MN;
        fo[15] = meanSq;
    }
}

// ---------------- kernel 2: per-feature batch stats (ddof=1) ----------------
__global__ __launch_bounds__(256) void colstats(const float* __restrict__ f,
                                                float* __restrict__ stats, int B) {
    __shared__ float ssum[256], ssq[256];
    const int tid  = threadIdx.x;
    const int col  = tid & 15;
    const int part = tid >> 4;           // 16 partials per column
    float s = 0.f, s2 = 0.f;
    for (int r = part; r < B; r += 16) {
        float v = f[(size_t)r * NFEAT + col];
        s += v; s2 += v * v;
    }
    ssum[tid] = s; ssq[tid] = s2;
    __syncthreads();
    if (part == 0) {
        float S = 0.f, S2 = 0.f;
        for (int p = 0; p < 16; ++p) { S += ssum[(p << 4) + col]; S2 += ssq[(p << 4) + col]; }
        float n    = (float)B;
        float mean = S / n;
        float var  = (S2 - n * mean * mean) / (n - 1.0f);
        var = var > 0.f ? var : 0.f;
        stats[col]      = mean;
        stats[16 + col] = 1.0f / (sqrtf(var) + 1e-6f);
    }
}

// ---------------- kernel 3: normalize + linear head via WMMA ----------------
// One wave32 per 16-row tile: D(16x16,f32) = A(16x32,f16) x B(32x16,f16),
// A = normalized features zero-padded K:16->32, B = w^T zero-padded N:2->16.
__global__ __launch_bounds__(32) void wmma_head(const float* __restrict__ f,
                                                const float* __restrict__ stats,
                                                const float* __restrict__ w,
                                                const float* __restrict__ bias,
                                                float* __restrict__ out) {
    const int tile = blockIdx.x;
    const int lane = threadIdx.x;      // 0..31, wave32, EXEC all ones
    const int ln   = lane & 15;
    const int hi   = lane >> 4;        // 0: lanes 0-15, 1: lanes 16-31
    const float* mean  = stats;
    const float* scale = stats + 16;

    // A-matrix layout (16-bit A 16x32, ISA 7.12.2): row M = ln for both halves;
    // element j -> K = (j<8 ? j : j+8) + (hi ? 8 : 0)
    v16h a;
    const int row = tile * 16 + ln;
    #pragma unroll
    for (int j = 0; j < 16; ++j) {
        int K = (j < 8 ? j : j + 8) + (hi ? 8 : 0);
        float v = 0.f;
        if (K < NFEAT)
            v = (f[(size_t)row * NFEAT + K] - mean[K]) * scale[K];
        a[j] = (_Float16)v;
    }

    // B-matrix layout (16-bit B 32x16): column N = ln; element j -> K = (j<8 ? j : j+8) + (hi ? 8 : 0)
    v16h bm;
    #pragma unroll
    for (int j = 0; j < 16; ++j) {
        int K = (j < 8 ? j : j + 8) + (hi ? 8 : 0);
        float v = 0.f;
        if (K < NFEAT && ln < 2) v = w[ln * NFEAT + K];   // B[K][N] = w[N][K]
        bm[j] = (_Float16)v;
    }

    v8f c = {};
    c = __builtin_amdgcn_wmma_f32_16x16x32_f16(
            /*neg_a=*/false, a, /*neg_b=*/false, bm,
            /*c_mod=*/(short)0, c, /*reuse_a=*/false, /*reuse_b=*/false);

    // C/D layout: VGPR r, lane L -> row r + (L<16?0:8), col L%16. Only cols 0,1 real.
    if (ln < 2) {
        const float bb = bias[ln];
        #pragma unroll
        for (int r = 0; r < 8; ++r) {
            int orow = tile * 16 + r + hi * 8;
            out[(size_t)orow * 2 + ln] = c[r] + bb;
        }
    }
}

extern "C" void kernel_launch(void* const* d_in, const int* in_sizes, int n_in,
                              void* d_out, int out_size, void* d_ws, size_t ws_size,
                              hipStream_t stream) {
    const float* x = (const float*)d_in[0];   // (B,1,3000) f32
    const float* w = (const float*)d_in[1];   // (2,16) f32
    const float* b = (const float*)d_in[2];   // (2,)  f32
    float* out = (float*)d_out;               // (B,2) f32

    const int B = in_sizes[0] / T_LEN;        // 8192

    float* f     = (float*)d_ws;              // B*16 features
    float* stats = f + (size_t)B * NFEAT;     // mean[16], invstd[16]

    feat_kernel<<<B, BLK, 0, stream>>>(x, f);
    colstats<<<1, 256, 0, stream>>>(f, stats, B);
    wmma_head<<<(B + 15) / 16, 32, 0, stream>>>(f, stats, w, b, out);
}
